// Model_22565758173396
// MI455X (gfx1250) — matrix operand order repaired and checked by reading, not compiled, not run on hardware
//
#include <hip/hip_runtime.h>
#include <hip/hip_bf16.h>

// ---------------------------------------------------------------------------
// Transformer encoder forward for MI455X (gfx1250, wave32, WMMA).
// GEMMs: v_wmma_f32_16x16x32_f16, block tile 128Mx64N, 8 waves/block.
// B tile (64x32 f16 = 4KB) staged into LDS with global_load_async_to_lds_b128
// (double-buffered, ASYNCcnt), shared by all 8 waves -> 8x less L2 traffic.
// B fragments are preloaded into registers so the 4 WMMAs issue back-to-back.
// ---------------------------------------------------------------------------

#define Bn 64
#define Sn 128
#define Dn 512
#define Hn 8
#define FFn 2048
#define Ln 6
#define DHn 64
#define BSn (Bn * Sn)          // 8192
#define BHn (Bn * Hn)          // 512

typedef _Float16 h8   __attribute__((ext_vector_type(8)));
typedef _Float16 v16h __attribute__((ext_vector_type(16)));
typedef float    v8f  __attribute__((ext_vector_type(8)));

#define SM_F32    0
#define SM_F16    1
#define SM_F16_VT 2   // scatter-store as per-head-transposed V: [BH, DH, S']

#define LDS_ROW 40            // 32 halves + 8 pad -> conflict-free ds_load_b128
#define LDS_BUF (64 * LDS_ROW)

// Async DMA of one 16B chunk: LDS[laddr] = *(Bt_base + goff_bytes)
__device__ __forceinline__ void async_b128_to_lds(unsigned laddr, unsigned goff,
                                                  const _Float16* base) {
  asm volatile("global_load_async_to_lds_b128 %0, %1, %2"
               :: "v"(laddr), "v"(goff), "s"(base) : "memory");
}

// ---------------------------------------------------------------------------
// WMMA GEMM: C[M,N] = A[M,K] * Bt[N,K]^T (+bias[n]) (+relu)
// A,Bt f16 row-major. Block = 8 waves = 128(M) x 64(N); grid exact (no tail).
// ---------------------------------------------------------------------------
template <int SMODE, bool RELU>
__global__ __launch_bounds__(256) void gemm_wmma(
    const _Float16* __restrict__ A, const _Float16* __restrict__ Bt,
    const float* __restrict__ bias, float* __restrict__ Cf,
    _Float16* __restrict__ Ch, int M, int N, int K,
    long sA, long sB, long sC) {
  __shared__ __align__(32) _Float16 bsh[2][LDS_BUF];

  const int z = blockIdx.z;
  A  += z * sA;
  Bt += z * sB;
  if (SMODE == SM_F32) Cf += z * sC; else Ch += z * sC;

  const int tilesN = N >> 6;
  const int m0 = (blockIdx.x / tilesN) * 128 + (threadIdx.x >> 5) * 16;
  const int n0 = (blockIdx.x % tilesN) * 64;

  const int lane = threadIdx.x & 31;
  const int rA   = lane & 15;                    // A row / B column within tile
  const int hi   = lane >> 4;                    // half-wave selector

  // A fragment source: row rA, K runs at hi*8 and hi*8+16 (ISA 16-bit A layout)
  const _Float16* Ap = A + (long)(m0 + rA) * K + hi * 8;

  // Cooperative B staging: thread t -> row t>>2 (of 64), 16B chunk t&3.
  const int srow = threadIdx.x >> 2, schk = threadIdx.x & 3;
  const unsigned lbase = (unsigned)(unsigned long long)(void*)&bsh[0][0];
  const unsigned lds0 = lbase + (unsigned)(srow * LDS_ROW + schk * 8) * 2;
  const unsigned lds1 = lds0 + LDS_BUF * 2;
  const unsigned gbase = ((unsigned)(n0 + srow) * (unsigned)K + schk * 8) * 2;

  v8f acc[4] = {{}, {}, {}, {}};

  async_b128_to_lds(lds0, gbase, Bt);            // prefetch k-step 0 -> buf 0
  for (int kk = 0; kk < K; kk += 32) {
    const int cur = (kk >> 5) & 1;
    if (kk + 32 < K) {
      async_b128_to_lds(cur ? lds0 : lds1, gbase + (unsigned)(kk + 32) * 2, Bt);
      asm volatile("s_wait_asynccnt 0x1" ::: "memory");   // cur buffer landed
    } else {
      asm volatile("s_wait_asynccnt 0x0" ::: "memory");
    }
    __syncthreads();

    union { v16h v; h8 h[2]; } af;
    af.h[0] = *(const h8*)(Ap + kk);
    af.h[1] = *(const h8*)(Ap + kk + 16);

    // Preload all four B fragments, then fire the WMMAs back-to-back so the
    // XDL pipe is not stalled on per-fragment s_wait_dscnt.
    const _Float16* lB = &bsh[cur][0];
    union { v16h v; h8 h[2]; } bf[4];
#pragma unroll
    for (int t = 0; t < 4; ++t) {
      const _Float16* p = lB + (t * 16 + rA) * LDS_ROW + hi * 8;
      bf[t].h[0] = *(const h8*)p;
      bf[t].h[1] = *(const h8*)(p + 16);
    }
#pragma unroll
    for (int t = 0; t < 4; ++t)
      acc[t] = __builtin_amdgcn_wmma_f32_16x16x32_f16(false, af.v, false, bf[t].v,
                                                      (short)0, acc[t], false, false);
    __syncthreads();                             // before cur buffer is re-filled
  }

  // Epilogue: C/D layout -> lane = N, VGPR r -> M = r + 8*(lane>=16)
  const int mBase = m0 + hi * 8;
#pragma unroll
  for (int t = 0; t < 4; ++t) {
    const int n = n0 + t * 16 + rA;
    const float bv = bias ? bias[n] : 0.0f;
#pragma unroll
    for (int r = 0; r < 8; ++r) {
      float v = acc[t][r] + bv;
      if (RELU) v = v > 0.0f ? v : 0.0f;
      const long flat = (long)(mBase + r) * N + n;
      if (SMODE == SM_F32) {
        Cf[flat] = v;
      } else if (SMODE == SM_F16) {
        Ch[flat] = (_Float16)v;
      } else {                       // per-head transpose of V (reshape quirk)
        const long bh = flat >> 13;  // / (S'*DH) = 8192
        const long pos = flat & 8191;
        const long s2 = pos >> 6, dh = pos & 63;
        Ch[bh * 8192 + dh * 128 + s2] = (_Float16)v;
      }
    }
  }
}

// ---------------------------------------------------------------------------
__global__ __launch_bounds__(256) void transpose_f32_to_f16(
    const float* __restrict__ W, _Float16* __restrict__ Wt, int K, int N) {
  const long i = (long)blockIdx.x * 256 + threadIdx.x;
  if (i >= (long)K * N) return;
  const long k = i / N, n = i % N;
  Wt[n * (long)K + k] = (_Float16)W[i];
}

__global__ __launch_bounds__(256) void embed_pe(
    const int* __restrict__ tok, const float* __restrict__ emb,
    float* __restrict__ xf, _Float16* __restrict__ xh) {
  const int bs = blockIdx.x;
  const int s = bs & (Sn - 1);
  const long tk = tok[bs];
  const float* er = emb + tk * (long)Dn;
  float* xr = xf + (long)bs * Dn;
  _Float16* xhr = xh + (long)bs * Dn;
  for (int d = threadIdx.x; d < Dn; d += 256) {
    const float i2 = (float)((d >> 1) << 1);
    const float ang = (float)s * expf(-(i2 * (1.0f / Dn)) * 9.210340371976184f);
    const float pe = (d & 1) ? cosf(ang) : sinf(ang);
    const float v = er[d] + pe;
    xr[d] = v;
    xhr[d] = (_Float16)v;
  }
}

__global__ __launch_bounds__(128) void softmax128(
    const float* __restrict__ S, _Float16* __restrict__ att) {
  __shared__ float red[128];
  const long row = blockIdx.x;
  const int t = threadIdx.x;
  const float v = S[row * 128 + t] * 0.125f;   // DH^-0.5
  red[t] = v; __syncthreads();
  for (int o = 64; o > 0; o >>= 1) { if (t < o) red[t] = fmaxf(red[t], red[t + o]); __syncthreads(); }
  const float m = red[0]; __syncthreads();
  const float e = expf(v - m);
  red[t] = e; __syncthreads();
  for (int o = 64; o > 0; o >>= 1) { if (t < o) red[t] += red[t + o]; __syncthreads(); }
  att[row * 128 + t] = (_Float16)(e / red[0]);
}

// out = LayerNorm(a + res) * g + b ; writes f32 + f16 copies
__global__ __launch_bounds__(256) void add_ln(
    const float* __restrict__ a, const float* __restrict__ res,
    const float* __restrict__ g, const float* __restrict__ b,
    float* __restrict__ of, _Float16* __restrict__ oh) {
  __shared__ float red[256];
  const long base = (long)blockIdx.x * Dn;
  const int t = threadIdx.x;
  const float v0 = a[base + t] + res[base + t];
  const float v1 = a[base + t + 256] + res[base + t + 256];
  red[t] = v0 + v1; __syncthreads();
  for (int o = 128; o > 0; o >>= 1) { if (t < o) red[t] += red[t + o]; __syncthreads(); }
  const float mean = red[0] * (1.0f / Dn); __syncthreads();
  const float d0 = v0 - mean, d1 = v1 - mean;
  red[t] = d0 * d0 + d1 * d1; __syncthreads();
  for (int o = 128; o > 0; o >>= 1) { if (t < o) red[t] += red[t + o]; __syncthreads(); }
  const float inv = rsqrtf(red[0] * (1.0f / Dn) + 1e-5f);
  const float r0 = d0 * inv * g[t] + b[t];
  const float r1 = d1 * inv * g[t + 256] + b[t + 256];
  of[base + t] = r0;        of[base + t + 256] = r1;
  oh[base + t] = (_Float16)r0; oh[base + t + 256] = (_Float16)r1;
}

// logits[b,c] = x[b,:] . Wc[:,c] + bc[c]  (K = S*D = 65536, tiny N: plain reduce)
__global__ __launch_bounds__(256) void classifier(
    const float* __restrict__ x, const float* __restrict__ Wc,
    const float* __restrict__ bc, float* __restrict__ out) {
  __shared__ float red[256];
  const int b = blockIdx.x / 10, c = blockIdx.x % 10;
  const float* xr = x + (long)b * (Sn * Dn);
  float acc = 0.0f;
  for (int k = threadIdx.x; k < Sn * Dn; k += 256) acc += xr[k] * Wc[(long)k * 10 + c];
  red[threadIdx.x] = acc; __syncthreads();
  for (int o = 128; o > 0; o >>= 1) { if (threadIdx.x < o) red[threadIdx.x] += red[threadIdx.x + o]; __syncthreads(); }
  if (threadIdx.x == 0) out[b * 10 + c] = red[0] + bc[c];
}

// ---------------------------------------------------------------------------
extern "C" void kernel_launch(void* const* d_in, const int* in_sizes, int n_in,
                              void* d_out, int out_size, void* d_ws, size_t ws_size,
                              hipStream_t stream) {
  const int*   tokens = (const int*)d_in[0];
  const float* emb    = (const float*)d_in[1];
  const float* Wq = (const float*)d_in[2],  *bq = (const float*)d_in[3];
  const float* Wk = (const float*)d_in[4],  *bk = (const float*)d_in[5];
  const float* Wv = (const float*)d_in[6],  *bv = (const float*)d_in[7];
  const float* Wo = (const float*)d_in[8],  *bo = (const float*)d_in[9];
  const float* ln1g = (const float*)d_in[10], *ln1b = (const float*)d_in[11];
  const float* W1 = (const float*)d_in[12], *b1 = (const float*)d_in[13];
  const float* W2 = (const float*)d_in[14], *b2 = (const float*)d_in[15];
  const float* ln2g = (const float*)d_in[16], *ln2b = (const float*)d_in[17];
  const float* Wc = (const float*)d_in[18], *bc = (const float*)d_in[19];
  float* out = (float*)d_out;

  // ---- workspace layout ----------------------------------------------------
  char* ws = (char*)d_ws;
  size_t off = 0;
  auto alloc = [&](size_t bytes) { char* p = ws + off; off += (bytes + 255) & ~(size_t)255; return p; };
  float*    xf32  = (float*)   alloc((size_t)BSn * Dn * 4);
  _Float16* xf16  = (_Float16*)alloc((size_t)BSn * Dn * 2);
  _Float16* wAll  = (_Float16*)alloc((size_t)Ln * (4 * Dn * Dn + 2 * Dn * FFn) * 2);
  _Float16* qf16  = (_Float16*)alloc((size_t)BSn * Dn * 2);
  _Float16* kf16  = (_Float16*)alloc((size_t)BSn * Dn * 2);
  _Float16* vt16  = (_Float16*)alloc((size_t)BSn * Dn * 2);
  char*     regA  = alloc((size_t)BHn * Sn * Sn * 4);        // scores f32 | h1 f16
  _Float16* att   = (_Float16*)alloc((size_t)BHn * Sn * Sn * 2);
  _Float16* ctx   = (_Float16*)alloc((size_t)BSn * Dn * 2);
  float*    t0    = (float*)   alloc((size_t)BSn * Dn * 4);
  float*    out32 = (float*)   alloc((size_t)BSn * Dn * 4);
  _Float16* out16 = (_Float16*)alloc((size_t)BSn * Dn * 2);
  float*    scores = (float*)regA;
  _Float16* h1     = (_Float16*)regA;
  (void)ws_size; (void)n_in; (void)in_sizes; (void)out_size;

  const long perL = 4L * Dn * Dn + 2L * Dn * FFn;

  // ---- weight prep: f32 -> f16, transposed to Bt[N][K] --------------------
  for (int l = 0; l < Ln; ++l) {
    _Float16* wl = wAll + l * perL;
    const long dd = (long)Dn * Dn, df = (long)Dn * FFn;
    transpose_f32_to_f16<<<(dd + 255) / 256, 256, 0, stream>>>(Wq + l * dd, wl,               Dn, Dn);
    transpose_f32_to_f16<<<(dd + 255) / 256, 256, 0, stream>>>(Wk + l * dd, wl + dd,          Dn, Dn);
    transpose_f32_to_f16<<<(dd + 255) / 256, 256, 0, stream>>>(Wv + l * dd, wl + 2 * dd,      Dn, Dn);
    transpose_f32_to_f16<<<(dd + 255) / 256, 256, 0, stream>>>(Wo + l * dd, wl + 3 * dd,      Dn, Dn);
    transpose_f32_to_f16<<<(df + 255) / 256, 256, 0, stream>>>(W1 + l * df, wl + 4 * dd,      Dn, FFn);
    transpose_f32_to_f16<<<(df + 255) / 256, 256, 0, stream>>>(W2 + l * df, wl + 4 * dd + df, FFn, Dn);
  }

  embed_pe<<<BSn, 256, 0, stream>>>(tokens, emb, xf32, xf16);

  // ---- encoder layers ------------------------------------------------------
  for (int l = 0; l < Ln; ++l) {
    _Float16* wl = wAll + l * perL;
    const long dd = (long)Dn * Dn, df = (long)Dn * FFn;
    const _Float16 *wqT = wl, *wkT = wl + dd, *wvT = wl + 2 * dd, *woT = wl + 3 * dd;
    const _Float16 *w1T = wl + 4 * dd, *w2T = wl + 4 * dd + df;

    // Q,K,V projections: M=8192,N=512 -> grid 64*8 = 512 blocks
    gemm_wmma<SM_F16, false><<<dim3(512, 1, 1), 256, 0, stream>>>(
        xf16, wqT, bq + l * Dn, nullptr, qf16, BSn, Dn, Dn, 0, 0, 0);
    gemm_wmma<SM_F16, false><<<dim3(512, 1, 1), 256, 0, stream>>>(
        xf16, wkT, bk + l * Dn, nullptr, kf16, BSn, Dn, Dn, 0, 0, 0);
    gemm_wmma<SM_F16_VT, false><<<dim3(512, 1, 1), 256, 0, stream>>>(
        xf16, wvT, bv + l * Dn, nullptr, vt16, BSn, Dn, Dn, 0, 0, 0);

    // scores = Q.K^T per head: z=512, M=N=128, K=64 (row-major K tile == Bt)
    gemm_wmma<SM_F32, false><<<dim3(2, 1, BHn), 256, 0, stream>>>(
        qf16, kf16, nullptr, scores, nullptr, Sn, Sn, DHn, 8192, 8192, 16384);

    softmax128<<<BHn * Sn, 128, 0, stream>>>(scores, att);

    // ctx = att.V: z=512, M=128, N=64, K=128 (Bt = transposed V)
    gemm_wmma<SM_F16, false><<<dim3(1, 1, BHn), 256, 0, stream>>>(
        att, vt16, nullptr, nullptr, ctx, Sn, DHn, Sn, 16384, 8192, 8192);

    // out-proj, residual + LN1
    gemm_wmma<SM_F32, false><<<dim3(512, 1, 1), 256, 0, stream>>>(
        ctx, woT, bo + l * Dn, t0, nullptr, BSn, Dn, Dn, 0, 0, 0);
    add_ln<<<BSn, 256, 0, stream>>>(t0, xf32, ln1g + l * Dn, ln1b + l * Dn, out32, out16);

    // FFN: relu(out@W1+b1) @ W2 + b2, residual + LN2 -> next x
    gemm_wmma<SM_F16, true><<<dim3(2048, 1, 1), 256, 0, stream>>>(
        out16, w1T, b1 + l * FFn, nullptr, h1, BSn, FFn, Dn, 0, 0, 0);
    gemm_wmma<SM_F32, false><<<dim3(512, 1, 1), 256, 0, stream>>>(
        h1, w2T, b2 + l * Dn, t0, nullptr, BSn, Dn, FFn, 0, 0, 0);
    add_ln<<<BSn, 256, 0, stream>>>(t0, out32, ln2g + l * Dn, ln2b + l * Dn, xf32, xf16);
  }

  classifier<<<Bn * 10, 256, 0, stream>>>(xf32, Wc, bc, out);
}
